// EqModelComplex_20143396619059
// MI455X (gfx1250) — matrix operand order repaired and checked by reading, not compiled, not run on hardware
//
#include <hip/hip_runtime.h>
#include <cmath>

typedef __attribute__((ext_vector_type(16))) _Float16 v16h;
typedef __attribute__((ext_vector_type(8)))  _Float16 v8h;
typedef __attribute__((ext_vector_type(8)))  float    v8f;

__device__ __forceinline__ v8f vzero() {
    v8f z = {0.f, 0.f, 0.f, 0.f, 0.f, 0.f, 0.f, 0.f};
    return z;
}

__device__ __forceinline__ v8f wmma(v16h a, v16h b, v8f c) {
    return __builtin_amdgcn_wmma_f32_16x16x32_f16(false, a, false, b, (short)0, c, false, false);
}

// A-fragment (16x32 f16): lane l holds row m0+(l&15); K halves {kb..kb+7, kb+16..kb+23}, kb = (l>>4)*8
__device__ __forceinline__ v16h load_afrag(const _Float16* __restrict__ A, int ld, int m0, int k0) {
    int l = threadIdx.x & 31;
    const _Float16* p = A + (size_t)(m0 + (l & 15)) * ld + k0 + ((l >> 4) << 3);
    v8h lo = *(const v8h*)p;
    v8h hi = *(const v8h*)(p + 16);
    return __builtin_shufflevector(lo, hi, 0, 1, 2, 3, 4, 5, 6, 7, 8, 9, 10, 11, 12, 13, 14, 15);
}

// B-fragment (32x16 f16) from a row-major (N,K) matrix: B[k][n] = W[n0+n][k0+k]
// lane l holds column n0+(l&15); K halves {kb..kb+7, kb+16..kb+23}, kb = (l>>4)*8
__device__ __forceinline__ v16h load_bfrag(const _Float16* __restrict__ W, int ld, int n0, int k0) {
    int l = threadIdx.x & 31;
    const _Float16* p = W + (size_t)(n0 + (l & 15)) * ld + k0 + ((l >> 4) << 3);
    v8h lo = *(const v8h*)p;
    v8h hi = *(const v8h*)(p + 16);
    return __builtin_shufflevector(lo, hi, 0, 1, 2, 3, 4, 5, 6, 7, 8, 9, 10, 11, 12, 13, 14, 15);
}

__device__ __forceinline__ float halfmax(float x) {
    x = fmaxf(x, __shfl_xor(x, 1));
    x = fmaxf(x, __shfl_xor(x, 2));
    x = fmaxf(x, __shfl_xor(x, 4));
    x = fmaxf(x, __shfl_xor(x, 8));
    return x;
}
__device__ __forceinline__ float halfsum(float x) {
    x += __shfl_xor(x, 1);
    x += __shfl_xor(x, 2);
    x += __shfl_xor(x, 4);
    x += __shfl_xor(x, 8);
    return x;
}

// ---------------- weight prep: W = exp(lm) * e^{i*ph} -> f16 wr/wi ----------------
__global__ void prep_w_kernel(const float* __restrict__ lm, const float* __restrict__ ph,
                              _Float16* __restrict__ wr, _Float16* __restrict__ wi, int n) {
    int i = blockIdx.x * blockDim.x + threadIdx.x;
    if (i < n) {
        float m = __expf(lm[i]);
        float p = ph[i];
        wr[i] = (_Float16)(m * __cosf(p));
        wi[i] = (_Float16)(m * __sinf(p));
    }
}

// ---------------- complex LayerNorm (row of D=512, 256 threads/block) ----------------
__device__ __forceinline__ float block_sum256(float v, float* red) {
    int t = threadIdx.x;
    red[t] = v;
    __syncthreads();
    for (int s = 128; s > 0; s >>= 1) {
        if (t < s) red[t] += red[t + s];
        __syncthreads();
    }
    float r = red[0];
    __syncthreads();
    return r;
}

__global__ void __launch_bounds__(256) cln_kernel(
    const float* __restrict__ xr, const float* __restrict__ xi,
    const float* __restrict__ gr, const float* __restrict__ gi,
    const float* __restrict__ br, const float* __restrict__ bi,
    _Float16* __restrict__ outr, _Float16* __restrict__ outi) {
    const int D = 512;
    __shared__ float red[256];
    int row = blockIdx.x;
    int t = threadIdx.x;
    const float* pr = xr + (size_t)row * D;
    const float* pi = xi + (size_t)row * D;
    float a0 = pr[t], a1 = pr[t + 256];
    float b0 = pi[t], b1 = pi[t + 256];
    float mr = block_sum256(a0 + a1, red) * (1.f / 512.f);
    float mi = block_sum256(b0 + b1, red) * (1.f / 512.f);
    float cr0 = a0 - mr, cr1 = a1 - mr, ci0 = b0 - mi, ci1 = b1 - mi;
    float var = block_sum256(cr0 * cr0 + ci0 * ci0 + cr1 * cr1 + ci1 * ci1, red) * (1.f / 512.f);
    float inv = rsqrtf(var + 1e-6f);
    float nr0 = cr0 * inv, ni0 = ci0 * inv;
    float nr1 = cr1 * inv, ni1 = ci1 * inv;
    size_t o = (size_t)row * D;
    outr[o + t]       = (_Float16)(nr0 * gr[t] - ni0 * gi[t] + br[t]);
    outi[o + t]       = (_Float16)(nr0 * gi[t] + ni0 * gr[t] + bi[t]);
    outr[o + t + 256] = (_Float16)(nr1 * gr[t + 256] - ni1 * gi[t + 256] + br[t + 256]);
    outi[o + t + 256] = (_Float16)(nr1 * gi[t + 256] + ni1 * gr[t + 256] + bi[t + 256]);
}

// ---------------- complex GEMM: out = A @ W^T (+bias)(+res); A:(M,K) W:(N,K) f16 ----------------
// Per k-step: all loads issued as one clause, then 16 WMMAs; next k-step A rows prefetched.
template <int NT>
__global__ void __launch_bounds__(256) cgemm_kernel(
    const _Float16* __restrict__ Ar, const _Float16* __restrict__ Ai,
    const _Float16* __restrict__ Wr, const _Float16* __restrict__ Wi,
    const float* __restrict__ bm, const float* __restrict__ bp,
    const float* __restrict__ resr, const float* __restrict__ resi,
    float* __restrict__ outr32, float* __restrict__ outi32,
    _Float16* __restrict__ outr16, _Float16* __restrict__ outi16,
    int M, int N, int K) {
    int l = threadIdx.x & 31;
    int lh = l >> 4, ln = l & 15;
    int wave = (blockIdx.x * blockDim.x + threadIdx.x) >> 5;
    int nwaves = (gridDim.x * blockDim.x) >> 5;
    int mt = M >> 4;
    int nt = N / (16 * NT);
    int ntiles = mt * nt;
    for (int tile = wave; tile < ntiles; tile += nwaves) {
        int tm = tile % mt, tn = tile / mt;
        int m0 = tm * 16, n0 = tn * 16 * NT;
        v8f accR[NT], accI[NT];
#pragma unroll
        for (int t = 0; t < NT; t++) { accR[t] = vzero(); accI[t] = vzero(); }
        for (int k0 = 0; k0 < K; k0 += 32) {
            // issue all fragment loads for this k-step up front (one VMEM clause)
            v16h arf = load_afrag(Ar, K, m0, k0);
            v16h aif = load_afrag(Ai, K, m0, k0);
            v16h wrf[NT], wif[NT];
#pragma unroll
            for (int t = 0; t < NT; t++) {
                wrf[t] = load_bfrag(Wr, K, n0 + t * 16, k0);
                wif[t] = load_bfrag(Wi, K, n0 + t * 16, k0);
            }
            // warm the next k-step's A rows into cache while WMMAs run
            if (k0 + 32 < K) {
                const _Float16* pa = Ar + (size_t)(m0 + (l & 15)) * K + k0 + 32;
                const _Float16* pb = Ai + (size_t)(m0 + (l & 15)) * K + k0 + 32;
                __builtin_prefetch(pa, 0, 3);
                __builtin_prefetch(pb, 0, 3);
            }
            v16h ainf = -aif;  // f16 WMMA has no A-neg modifier; negate fragment
#pragma unroll
            for (int t = 0; t < NT; t++) {
                accR[t] = wmma(arf,  wrf[t], accR[t]);
                accR[t] = wmma(ainf, wif[t], accR[t]);
                accI[t] = wmma(arf,  wif[t], accI[t]);
                accI[t] = wmma(aif,  wrf[t], accI[t]);
            }
        }
#pragma unroll
        for (int t = 0; t < NT; t++) {
            int col = n0 + t * 16 + ln;
            float biasr = 0.f, biasi = 0.f;
            if (bm) {
                float m_ = bm[col], p_ = bp[col];
                biasr = m_ * __cosf(p_);
                biasi = m_ * __sinf(p_);
            }
#pragma unroll
            for (int v = 0; v < 8; v++) {
                int row = m0 + v + lh * 8;
                size_t o = (size_t)row * N + col;
                float r  = accR[t][v] + biasr;
                float im = accI[t][v] + biasi;
                if (resr) { r += resr[o]; im += resi[o]; }
                if (outr32) { outr32[o] = r; outi32[o] = im; }
                if (outr16) { outr16[o] = (_Float16)r; outi16[o] = (_Float16)im; }
            }
        }
    }
}

// ---------------- RoPE on q,k (in place, f16) + V transpose to (B,H,HD,S) ----------------
__global__ void __launch_bounds__(256) rope_v_kernel(
    _Float16* __restrict__ qr, _Float16* __restrict__ qi,
    _Float16* __restrict__ kr, _Float16* __restrict__ ki,
    const _Float16* __restrict__ vr, const _Float16* __restrict__ vi,
    _Float16* __restrict__ vtr, _Float16* __restrict__ vti) {
    const int S = 1024, D = 512, H = 8, HD = 64;
    int idx = blockIdx.x * blockDim.x + threadIdx.x;  // B*S*H*32 = 1048576
    if (idx >= 4 * S * H * 32) return;
    int j = idx & 31;
    int h = (idx >> 5) & 7;
    int s = (idx >> 8) & 1023;
    int b = idx >> 18;
    float invf = __powf(10000.f, -(float)(2 * j) / (float)HD);
    float ang = (float)s * invf;
    float c = __cosf(ang), sn = __sinf(ang);
    size_t base = ((size_t)(b * S + s)) * D + h * HD;
    int d0 = j, d1 = j + 32;
#define ROT(P)                                               \
    {                                                        \
        float x0 = (float)P[base + d0];                      \
        float x1 = (float)P[base + d1];                      \
        P[base + d0] = (_Float16)(x0 * c - x1 * sn);         \
        P[base + d1] = (_Float16)(x1 * c + x0 * sn);         \
    }
    ROT(qr) ROT(qi) ROT(kr) ROT(ki)
#undef ROT
    size_t tb = (size_t)(b * H + h) * HD;
    vtr[(tb + d0) * S + s] = vr[base + d0];
    vtr[(tb + d1) * S + s] = vr[base + d1];
    vti[(tb + d0) * S + s] = vi[base + d0];
    vti[(tb + d1) * S + s] = vi[base + d1];
}

// ---------------- flash-style Hermitian attention: 1 wave per 16-row q-tile ----------------
__global__ void __launch_bounds__(32) attn_kernel(
    const _Float16* __restrict__ qr, const _Float16* __restrict__ qi,
    const _Float16* __restrict__ kr, const _Float16* __restrict__ ki,
    const _Float16* __restrict__ vtr, const _Float16* __restrict__ vti,
    _Float16* __restrict__ ar, _Float16* __restrict__ ai) {
    const int S = 1024, D = 512, H = 8, HD = 64;
    const float scale = 0.125f;  // 1/sqrt(64)
    int bid = blockIdx.x;
    int qt = bid & 63;
    int h = (bid >> 6) & 7;
    int b = bid >> 9;
    int q0 = qt * 16;
    int l = threadIdx.x & 31;
    int lh = l >> 4, ln = l & 15;

    __shared__ alignas(32) _Float16 pbuf[16 * 32];

    const _Float16* qrB = qr + (size_t)b * S * D + h * HD;
    const _Float16* qiB = qi + (size_t)b * S * D + h * HD;
    const _Float16* krB = kr + (size_t)b * S * D + h * HD;
    const _Float16* kiB = ki + (size_t)b * S * D + h * HD;
    const _Float16* vtrB = vtr + (size_t)(b * H + h) * HD * S;
    const _Float16* vtiB = vti + (size_t)(b * H + h) * HD * S;

    v16h qrf0 = load_afrag(qrB, D, q0, 0);
    v16h qrf1 = load_afrag(qrB, D, q0, 32);
    v16h qif0 = load_afrag(qiB, D, q0, 0);
    v16h qif1 = load_afrag(qiB, D, q0, 32);

    v8f outr_[4], outi_[4];
    float mrow[8], lrow[8];
#pragma unroll
    for (int t = 0; t < 4; t++) { outr_[t] = vzero(); outi_[t] = vzero(); }
#pragma unroll
    for (int v = 0; v < 8; v++) { mrow[v] = -3.0e38f; lrow[v] = 0.f; }

    for (int kt = 0; kt < q0 + 16; kt += 32) {
        // --- 1) load all K^T fragments for this 32-key chunk (one clause) ---
        v16h bkr[2][2], bki[2][2];
#pragma unroll
        for (int j = 0; j < 2; j++) {
            int key0 = kt + 16 * j;
            bkr[j][0] = load_bfrag(krB, D, key0, 0);
            bkr[j][1] = load_bfrag(krB, D, key0, 32);
            bki[j][0] = load_bfrag(kiB, D, key0, 0);
            bki[j][1] = load_bfrag(kiB, D, key0, 32);
        }
        // --- 2) score WMMAs: Re(q . conj(k)) = qr.kr + qi.ki ---
        v8f sc[2];
#pragma unroll
        for (int j = 0; j < 2; j++) {
            v8f s0 = vzero();
            s0 = wmma(qrf0, bkr[j][0], s0);
            s0 = wmma(qrf1, bkr[j][1], s0);
            s0 = wmma(qif0, bki[j][0], s0);
            s0 = wmma(qif1, bki[j][1], s0);
            sc[j] = s0;
        }
        // --- 3) issue V fragment loads NOW; softmax VALU work hides their latency ---
        v16h bvr[4], bvi[4];
#pragma unroll
        for (int t = 0; t < 4; t++) {
            bvr[t] = load_bfrag(vtrB, S, t * 16, kt);
            bvi[t] = load_bfrag(vtiB, S, t * 16, kt);
        }
        // --- 4) causal mask + online softmax (16-lane butterfly reductions) ---
        float alpha[8];
#pragma unroll
        for (int v = 0; v < 8; v++) {
            int row = q0 + v + lh * 8;
            float e0 = sc[0][v] * scale;
            float e1 = sc[1][v] * scale;
            if (kt + ln > row) e0 = -1.0e9f;
            if (kt + 16 + ln > row) e1 = -1.0e9f;
            float m = halfmax(fmaxf(e0, e1));
            m = fmaxf(m, mrow[v]);
            float a = __expf(mrow[v] - m);
            float p0 = __expf(e0 - m);
            float p1 = __expf(e1 - m);
            float ps = halfsum(p0 + p1);
            pbuf[(v + lh * 8) * 32 + ln] = (_Float16)p0;
            pbuf[(v + lh * 8) * 32 + 16 + ln] = (_Float16)p1;
            alpha[v] = a;
            mrow[v] = m;
            lrow[v] = lrow[v] * a + ps;
        }
#pragma unroll
        for (int t = 0; t < 4; t++)
#pragma unroll
            for (int v = 0; v < 8; v++) {
                outr_[t][v] *= alpha[v];
                outi_[t][v] *= alpha[v];
            }
        __builtin_amdgcn_wave_barrier();
        asm volatile("s_wait_dscnt 0" ::: "memory");
        // --- 5) P fragment (16x32 A-layout) from LDS, then PV WMMAs ---
        const _Float16* pp = &pbuf[ln * 32 + (lh << 3)];
        v8h lo = *(const v8h*)pp;
        v8h hi = *(const v8h*)(pp + 16);
        v16h pf = __builtin_shufflevector(lo, hi, 0, 1, 2, 3, 4, 5, 6, 7, 8, 9, 10, 11, 12, 13, 14, 15);
#pragma unroll
        for (int t = 0; t < 4; t++) {
            outr_[t] = wmma(pf, bvr[t], outr_[t]);
            outi_[t] = wmma(pf, bvi[t], outi_[t]);
        }
        __builtin_amdgcn_wave_barrier();
    }
#pragma unroll
    for (int v = 0; v < 8; v++) {
        float inv = 1.0f / lrow[v];
        int row = q0 + v + lh * 8;
        size_t o = ((size_t)(b * S) + row) * D + h * HD + ln;
#pragma unroll
        for (int t = 0; t < 4; t++) {
            ar[o + t * 16] = (_Float16)(outr_[t][v] * inv);
            ai[o + t * 16] = (_Float16)(outi_[t][v] * inv);
        }
    }
}

// ---------------- fused gate+up complex GEMMs with |g|-sigmoid complex gating ----------------
__global__ void __launch_bounds__(256) gateup_kernel(
    const _Float16* __restrict__ Ar, const _Float16* __restrict__ Ai,
    const _Float16* __restrict__ Gr, const _Float16* __restrict__ Gi,
    const _Float16* __restrict__ Ur, const _Float16* __restrict__ Ui,
    _Float16* __restrict__ hidr, _Float16* __restrict__ hidi,
    int M, int N, int K) {
    int l = threadIdx.x & 31;
    int lh = l >> 4, ln = l & 15;
    int wave = (blockIdx.x * blockDim.x + threadIdx.x) >> 5;
    int nwaves = (gridDim.x * blockDim.x) >> 5;
    int mt = M >> 4;
    int ntiles = mt * (N >> 4);
    for (int tile = wave; tile < ntiles; tile += nwaves) {
        int m0 = (tile % mt) * 16, n0 = (tile / mt) * 16;
        v8f gR = vzero(), gI = vzero(), uR = vzero(), uI = vzero();
        for (int k0 = 0; k0 < K; k0 += 32) {
            // group all loads of this k-step before the WMMA burst
            v16h arf = load_afrag(Ar, K, m0, k0);
            v16h aif = load_afrag(Ai, K, m0, k0);
            v16h grf = load_bfrag(Gr, K, n0, k0);
            v16h gif = load_bfrag(Gi, K, n0, k0);
            v16h urf = load_bfrag(Ur, K, n0, k0);
            v16h uif = load_bfrag(Ui, K, n0, k0);
            if (k0 + 32 < K) {
                __builtin_prefetch(Ar + (size_t)(m0 + (l & 15)) * K + k0 + 32, 0, 3);
                __builtin_prefetch(Ai + (size_t)(m0 + (l & 15)) * K + k0 + 32, 0, 3);
            }
            v16h ainf = -aif;
            gR = wmma(arf, grf, gR);  gR = wmma(ainf, gif, gR);
            gI = wmma(arf, gif, gI);  gI = wmma(aif,  grf, gI);
            uR = wmma(arf, urf, uR);  uR = wmma(ainf, uif, uR);
            uI = wmma(arf, uif, uI);  uI = wmma(aif,  urf, uI);
        }
        int col = n0 + ln;
#pragma unroll
        for (int v = 0; v < 8; v++) {
            float gr_ = gR[v], gi_ = gI[v], ur_ = uR[v], ui_ = uI[v];
            float mag = sqrtf(gr_ * gr_ + gi_ * gi_);
            float sg = 1.f / (1.f + __expf(-mag));
            float gar = gr_ * sg, gai = gi_ * sg;
            size_t o = (size_t)(m0 + v + lh * 8) * N + col;
            hidr[o] = (_Float16)(gar * ur_ - gai * ui_);
            hidi[o] = (_Float16)(gar * ui_ + gai * ur_);
        }
    }
}

// ---------------- host launcher ----------------
extern "C" void kernel_launch(void* const* d_in, const int* in_sizes, int n_in,
                              void* d_out, int out_size, void* d_ws, size_t ws_size,
                              hipStream_t stream) {
    (void)in_sizes; (void)n_in; (void)out_size; (void)ws_size;
    const int B = 4, S = 1024, D = 512, H = 8, HID = 2048;
    const int M = B * S;  // 4096

    const float* x_r = (const float*)d_in[0];
    const float* x_i = (const float*)d_in[1];
    const float* ln1_gr = (const float*)d_in[2];
    const float* ln1_gi = (const float*)d_in[3];
    const float* ln1_br = (const float*)d_in[4];
    const float* ln1_bi = (const float*)d_in[5];
    const float* q_lm = (const float*)d_in[6];
    const float* q_ph = (const float*)d_in[7];
    const float* q_bm = (const float*)d_in[8];
    const float* q_bp = (const float*)d_in[9];
    const float* k_lm = (const float*)d_in[10];
    const float* k_ph = (const float*)d_in[11];
    const float* k_bm = (const float*)d_in[12];
    const float* k_bp = (const float*)d_in[13];
    const float* v_lm = (const float*)d_in[14];
    const float* v_ph = (const float*)d_in[15];
    const float* v_bm = (const float*)d_in[16];
    const float* v_bp = (const float*)d_in[17];
    const float* o_lm = (const float*)d_in[18];
    const float* o_ph = (const float*)d_in[19];
    const float* o_bm = (const float*)d_in[20];
    const float* o_bp = (const float*)d_in[21];
    const float* ln2_gr = (const float*)d_in[22];
    const float* ln2_gi = (const float*)d_in[23];
    const float* ln2_br = (const float*)d_in[24];
    const float* ln2_bi = (const float*)d_in[25];
    const float* gate_lm = (const float*)d_in[26];
    const float* gate_ph = (const float*)d_in[27];
    const float* up_lm = (const float*)d_in[28];
    const float* up_ph = (const float*)d_in[29];
    const float* down_lm = (const float*)d_in[30];
    const float* down_ph = (const float*)d_in[31];

    char* base = (char*)d_ws;
    size_t off = 0;
    auto alloc = [&](size_t nbytes) -> void* {
        void* p = base + off;
        off = (off + nbytes + 255) & ~(size_t)255;
        return p;
    };
    const size_t h2 = sizeof(_Float16);
    _Float16* Wq_r = (_Float16*)alloc((size_t)D * D * h2);
    _Float16* Wq_i = (_Float16*)alloc((size_t)D * D * h2);
    _Float16* Wk_r = (_Float16*)alloc((size_t)D * D * h2);
    _Float16* Wk_i = (_Float16*)alloc((size_t)D * D * h2);
    _Float16* Wv_r = (_Float16*)alloc((size_t)D * D * h2);
    _Float16* Wv_i = (_Float16*)alloc((size_t)D * D * h2);
    _Float16* Wo_r = (_Float16*)alloc((size_t)D * D * h2);
    _Float16* Wo_i = (_Float16*)alloc((size_t)D * D * h2);
    _Float16* Wg_r = (_Float16*)alloc((size_t)HID * D * h2);
    _Float16* Wg_i = (_Float16*)alloc((size_t)HID * D * h2);
    _Float16* Wu_r = (_Float16*)alloc((size_t)HID * D * h2);
    _Float16* Wu_i = (_Float16*)alloc((size_t)HID * D * h2);
    _Float16* Wd_r = (_Float16*)alloc((size_t)D * HID * h2);
    _Float16* Wd_i = (_Float16*)alloc((size_t)D * HID * h2);
    _Float16* h_r = (_Float16*)alloc((size_t)M * D * h2);
    _Float16* h_i = (_Float16*)alloc((size_t)M * D * h2);
    _Float16* q_r = (_Float16*)alloc((size_t)M * D * h2);
    _Float16* q_i = (_Float16*)alloc((size_t)M * D * h2);
    _Float16* k_r = (_Float16*)alloc((size_t)M * D * h2);
    _Float16* k_i = (_Float16*)alloc((size_t)M * D * h2);
    _Float16* v_r = (_Float16*)alloc((size_t)M * D * h2);
    _Float16* v_i = (_Float16*)alloc((size_t)M * D * h2);
    _Float16* vt_r = (_Float16*)alloc((size_t)M * D * h2);
    _Float16* vt_i = (_Float16*)alloc((size_t)M * D * h2);
    _Float16* a_r = (_Float16*)alloc((size_t)M * D * h2);
    _Float16* a_i = (_Float16*)alloc((size_t)M * D * h2);
    _Float16* hid_r = (_Float16*)alloc((size_t)M * HID * h2);
    _Float16* hid_i = (_Float16*)alloc((size_t)M * HID * h2);
    float* res_r = (float*)alloc((size_t)M * D * sizeof(float));
    float* res_i = (float*)alloc((size_t)M * D * sizeof(float));

    float* out_r = (float*)d_out;
    float* out_i = out_r + (size_t)M * D;

    // 1) weights
    prep_w_kernel<<<(D * D + 255) / 256, 256, 0, stream>>>(q_lm, q_ph, Wq_r, Wq_i, D * D);
    prep_w_kernel<<<(D * D + 255) / 256, 256, 0, stream>>>(k_lm, k_ph, Wk_r, Wk_i, D * D);
    prep_w_kernel<<<(D * D + 255) / 256, 256, 0, stream>>>(v_lm, v_ph, Wv_r, Wv_i, D * D);
    prep_w_kernel<<<(D * D + 255) / 256, 256, 0, stream>>>(o_lm, o_ph, Wo_r, Wo_i, D * D);
    prep_w_kernel<<<(HID * D + 255) / 256, 256, 0, stream>>>(gate_lm, gate_ph, Wg_r, Wg_i, HID * D);
    prep_w_kernel<<<(HID * D + 255) / 256, 256, 0, stream>>>(up_lm, up_ph, Wu_r, Wu_i, HID * D);
    prep_w_kernel<<<(D * HID + 255) / 256, 256, 0, stream>>>(down_lm, down_ph, Wd_r, Wd_i, D * HID);

    // 2) LN1
    cln_kernel<<<M, 256, 0, stream>>>(x_r, x_i, ln1_gr, ln1_gi, ln1_br, ln1_bi, h_r, h_i);

    // 3) QKV projections (each: 2048 wave-tiles -> 256 blocks of 8 waves)
    cgemm_kernel<4><<<256, 256, 0, stream>>>(h_r, h_i, Wq_r, Wq_i, q_bm, q_bp,
                                             nullptr, nullptr, nullptr, nullptr, q_r, q_i, M, D, D);
    cgemm_kernel<4><<<256, 256, 0, stream>>>(h_r, h_i, Wk_r, Wk_i, k_bm, k_bp,
                                             nullptr, nullptr, nullptr, nullptr, k_r, k_i, M, D, D);
    cgemm_kernel<4><<<256, 256, 0, stream>>>(h_r, h_i, Wv_r, Wv_i, v_bm, v_bp,
                                             nullptr, nullptr, nullptr, nullptr, v_r, v_i, M, D, D);

    // 4) RoPE (q,k in place) + V transpose to (B,H,HD,S)
    rope_v_kernel<<<4096, 256, 0, stream>>>(q_r, q_i, k_r, k_i, v_r, v_i, vt_r, vt_i);

    // 5) attention: B*H*(S/16) = 2048 q-tiles, 1 wave each
    attn_kernel<<<2048, 32, 0, stream>>>(q_r, q_i, k_r, k_i, vt_r, vt_i, a_r, a_i);

    // 6) O projection + residual -> res (f32)
    cgemm_kernel<4><<<256, 256, 0, stream>>>(a_r, a_i, Wo_r, Wo_i, o_bm, o_bp,
                                             x_r, x_i, res_r, res_i, nullptr, nullptr, M, D, D);

    // 7) LN2 -> f16
    cln_kernel<<<M, 256, 0, stream>>>(res_r, res_i, ln2_gr, ln2_gi, ln2_br, ln2_bi, h_r, h_i);

    // 8) fused gate/up + complex SwiGLU gating -> hid (f16)
    gateup_kernel<<<4096, 256, 0, stream>>>(h_r, h_i, Wg_r, Wg_i, Wu_r, Wu_i, hid_r, hid_i, M, HID, D);

    // 9) down projection + residual -> d_out (f32, [out_r | out_i])
    cgemm_kernel<4><<<256, 256, 0, stream>>>(hid_r, hid_i, Wd_r, Wd_i, nullptr, nullptr,
                                             res_r, res_i, out_r, out_i, nullptr, nullptr, M, D, HID);
}
